// EllipseAreaNetwork_31782757990482
// MI455X (gfx1250) — matrix operand order repaired and checked by reading, not compiled, not run on hardware
//
#include <hip/hip_runtime.h>
#include <hip/hip_bf16.h>
#include <math.h>

// ---------------------------------------------------------------------------
// EllipseAreaNetwork forward on gfx1250 (MI455X).
// Dense MLP GEMMs use v_wmma_f32_16x16x32_f16 (wave32 WMMA); the per-edge
// feature gather uses GLOBAL_LOAD_ASYNC_TO_LDS_B128 (ASYNCcnt-tracked);
// scatter aggregation uses ordered-float atomic max; k-means runs a fixed
// 50 iterations.
// ---------------------------------------------------------------------------

typedef _Float16 f16;
typedef __attribute__((ext_vector_type(16))) _Float16 v16h;
typedef __attribute__((ext_vector_type(8)))  _Float16 v8h;
typedef __attribute__((ext_vector_type(8)))  float    v8f;

#define KCLUST 8
#define NBATCH 16
#define USE_ASYNC_LDS 1

// ---------------------------------------------------------------------- utils

__device__ __forceinline__ float softplusf(float x) {
    return x > 20.f ? x : log1pf(expf(x));
}

// Ordered-float atomic max (works for any sign; buffer init to -inf).
__device__ __forceinline__ void atomicMaxF32(float* addr, float v) {
    if (v >= 0.f) atomicMax((int*)addr, __float_as_int(v));
    else          atomicMin((unsigned int*)addr, __float_as_uint(v));
}

#if USE_ASYNC_LDS
// Low 32 bits of a generic shared-memory pointer are the LDS byte offset
// (ISA 10.2: LDS aperture address = {SHARED_BASE, addr[31:0]}).
__device__ __forceinline__ unsigned lds_off(const void* p) {
    return (unsigned)(size_t)p;
}
__device__ __forceinline__ void async_copy_b128(unsigned ldsOff, const void* gptr) {
    asm volatile("global_load_async_to_lds_b128 %0, %1, off"
                 :: "v"(ldsOff), "v"((unsigned long long)(size_t)gptr)
                 : "memory");
}
__device__ __forceinline__ void wait_async0() {
#if __has_builtin(__builtin_amdgcn_s_wait_asynccnt)
    __builtin_amdgcn_s_wait_asynccnt(0);
#else
    asm volatile("s_wait_asynccnt 0x0" ::: "memory");
#endif
}
#endif

// A-fragment 16x32 f16 from an LDS row: per lane, elements 0..7 are halves
// [kBase + laneHi*8, +8) and elements 8..15 are [kBase + 16 + laneHi*8, +8):
// two contiguous 16B LDS loads.
__device__ __forceinline__ v16h load_A_lds(const f16* rowBase, int kBase, int laneHi) {
    const f16* p = rowBase + kBase + laneHi * 8;
    v8h lo = *(const v8h*)p;
    v8h hi = *(const v8h*)(p + 16);
    return __builtin_shufflevector(lo, hi, 0, 1, 2, 3, 4, 5, 6, 7,
                                           8, 9, 10, 11, 12, 13, 14, 15);
}

// Same, but value = xj[k] - xi[k] (the second half of the EdgeConv feature).
__device__ __forceinline__ v16h load_A_diff_lds(const f16* xjRow, const f16* xiRow,
                                                int kBase, int laneHi) {
    int o = kBase + laneHi * 8;
    v8h lo = *(const v8h*)(xjRow + o)      - *(const v8h*)(xiRow + o);
    v8h hi = *(const v8h*)(xjRow + o + 16) - *(const v8h*)(xiRow + o + 16);
    return __builtin_shufflevector(lo, hi, 0, 1, 2, 3, 4, 5, 6, 7,
                                           8, 9, 10, 11, 12, 13, 14, 15);
}

// B-fragment 32x16 f16 from row-major W[NOUT][K] (B = W^T tile): per lane,
// elements 0..7 are [kBase + laneHi*8, +8) and 8..15 are [kBase + 16 + laneHi*8, +8).
__device__ __forceinline__ v16h load_B_W(const f16* W, int K, int nBase, int kBase, int lane) {
    const f16* wrow = W + (size_t)(nBase + (lane & 15)) * K + kBase + ((lane >> 4) << 3);
    v8h lo = *(const v8h*)wrow;
    v8h hi = *(const v8h*)(wrow + 16);
    return __builtin_shufflevector(lo, hi, 0, 1, 2, 3, 4, 5, 6, 7,
                                           8, 9, 10, 11, 12, 13, 14, 15);
}

// ------------------------------------------------------------------ small ops

__global__ void fill_f32(float* p, float v, int n) {
    int i = blockIdx.x * blockDim.x + threadIdx.x;
    if (i < n) p[i] = v;
}

__global__ void f32_to_f16_kernel(const float* __restrict__ in, f16* __restrict__ out, int n) {
    int i = blockIdx.x * blockDim.x + threadIdx.x;
    if (i < n) out[i] = (f16)in[i];
}

// replace +-inf with 0 (empty-segment convention), optional relu
__global__ void fixup_kernel(float* p, int n, int relu) {
    int i = blockIdx.x * blockDim.x + threadIdx.x;
    if (i >= n) return;
    float v = p[i];
    if (!__builtin_isfinite(v)) v = 0.f;
    if (relu) v = fmaxf(v, 0.f);
    p[i] = v;
}

__global__ void deg_kernel(const int* __restrict__ src, float* __restrict__ deg, int E) {
    int e = blockIdx.x * blockDim.x + threadIdx.x;
    if (e < E) atomicAdd(&deg[src[e]], 1.f);
}

// conv1 node linears (K too small for WMMA): out = relu(X_cat @ W^T + b)
// input column c maps to X[:, c % KIN] (duplication implements cat([x, x])).
template<int K, int NOUT>
__global__ void small_linear_kernel(const float* __restrict__ X, int KIN,
                                    const float* __restrict__ W, const float* __restrict__ b,
                                    float* __restrict__ outF, f16* __restrict__ outH, int M) {
    int idx = blockIdx.x * blockDim.x + threadIdx.x;
    int i = idx / NOUT, o = idx - i * NOUT;
    if (i >= M) return;
    float acc = b[o];
#pragma unroll
    for (int c = 0; c < K; ++c)
        acc += X[(size_t)i * KIN + (c % KIN)] * W[o * K + c];
    acc = fmaxf(acc, 0.f);
    if (outF) outF[(size_t)i * NOUT + o] = acc;
    if (outH) outH[(size_t)i * NOUT + o] = (f16)acc;
}

// h1 (f32 [M,64]) -> h1h (f16 [M,64]) and hcat (f16 [M,128] = cat([h1,h1]))
__global__ void h_to_half_cat(const float* __restrict__ h, f16* __restrict__ hh,
                              f16* __restrict__ hcat, int M) {
    int idx = blockIdx.x * blockDim.x + threadIdx.x;
    int i = idx >> 6, c = idx & 63;
    if (i >= M) return;
    f16 v = (f16)h[(size_t)i * 64 + c];
    hh[(size_t)i * 64 + c] = v;
    hcat[(size_t)i * 128 + c] = v;
    hcat[(size_t)i * 128 + 64 + c] = v;
}

// --------------------------------------------------------------- WMMA GEMMs

// out = act(A[M,K]f16 @ W[NOUT,K]^T + b); one wave per 16-row tile.
template<int K, int NOUT, bool RELU>
__global__ void __launch_bounds__(32) gemm_wmma_kernel(
        const f16* __restrict__ A, const f16* __restrict__ W,
        const float* __restrict__ bias,
        float* __restrict__ outF, f16* __restrict__ outH, int M) {
    __shared__ alignas(16) f16 ldsA[16 * K];
    const int lane = threadIdx.x;
    const int r0 = blockIdx.x * 16;
    constexpr int CH8 = K / 8;                       // 16B chunks per row
    for (int ch = lane; ch < 16 * CH8; ch += 32) {
        int r = ch / CH8, c8 = (ch - r * CH8) * 8;
        int rr = r0 + r; if (rr >= M) rr = M - 1;
        *(v8h*)&ldsA[r * K + c8] = *(const v8h*)&A[(size_t)rr * K + c8];
    }
    __syncthreads();
    const int laneHi = lane >> 4;
    const int row = lane & 15;
    constexpr int NT = NOUT / 16;
    v8f acc[NT];
#pragma unroll
    for (int t = 0; t < NT; ++t) { v8f z = {}; acc[t] = z; }
#pragma unroll
    for (int kb = 0; kb < K; kb += 32) {
        v16h a = load_A_lds(&ldsA[row * K], kb, laneHi);
#pragma unroll
        for (int t = 0; t < NT; ++t) {
            v16h b = load_B_W(W, K, t * 16, kb, lane);
            acc[t] = __builtin_amdgcn_wmma_f32_16x16x32_f16(
                false, a, false, b, (short)0, acc[t], false, false);
        }
    }
#pragma unroll
    for (int t = 0; t < NT; ++t) {
        int n = t * 16 + (lane & 15);
        float bv = bias[n];
#pragma unroll
        for (int v = 0; v < 8; ++v) {
            int r = r0 + v + laneHi * 8;
            if (r < M) {
                float val = acc[t][v] + bv;
                if (RELU) val = fmaxf(val, 0.f);
                if (outF) outF[(size_t)r * NOUT + n] = val;
                if (outH) outH[(size_t)r * NOUT + n] = (f16)val;
            }
        }
    }
}

// EdgeConv: m_e = cat([x_i, x_j - x_i]) (dim 2C), relu(m @ W1^T + b1) @ W2^T + b2,
// scatter-max into out[dst] (64 cols). One wave handles 16 edges. The raw x_i /
// x_j rows are copied to LDS with async b128 copies; the cat/diff is folded
// into A-fragment construction (each 32-wide K-step falls entirely in one half
// since C % 32 == 0).
template<int C>
__global__ void __launch_bounds__(32) edgeconv_wmma_kernel(
        const f16* __restrict__ sb, const int* __restrict__ src, const int* __restrict__ dst,
        const f16* __restrict__ W1, const float* __restrict__ b1,   // [64, 2C], [64]
        const f16* __restrict__ W2, const float* __restrict__ b2,   // [64, 64], [64]
        float* __restrict__ out, int nEdges) {
    constexpr int KA = 2 * C;
    __shared__ alignas(16) f16 ldsXi[16 * C];
    __shared__ alignas(16) f16 ldsXj[16 * C];
    __shared__ alignas(16) f16 ldsH[16 * 64];
    const int lane = threadIdx.x;
    const int e0 = blockIdx.x * 16;

    // stage x_i (= sb[dst]) and x_j (= sb[src]) rows into LDS, 16B per op
    constexpr int CH8 = C / 8;
    for (int ch = lane; ch < 16 * CH8; ch += 32) {
        int er = ch / CH8, c8 = (ch - er * CH8) * 8;
        int e = e0 + er; if (e >= nEdges) e = nEdges - 1;
        int d = dst[e], s = src[e];
        const f16* gi = sb + (size_t)d * C + c8;
        const f16* gj = sb + (size_t)s * C + c8;
#if USE_ASYNC_LDS
        async_copy_b128(lds_off(&ldsXi[er * C + c8]), gi);
        async_copy_b128(lds_off(&ldsXj[er * C + c8]), gj);
#else
        *(v8h*)&ldsXi[er * C + c8] = *(const v8h*)gi;
        *(v8h*)&ldsXj[er * C + c8] = *(const v8h*)gj;
#endif
    }
#if USE_ASYNC_LDS
    wait_async0();
#endif
    __syncthreads();

    const int laneHi = lane >> 4;
    const int row = lane & 15;

    // GEMM1: [16 x 2C] x [2C x 64]
    v8f acc[4];
#pragma unroll
    for (int t = 0; t < 4; ++t) { v8f z = {}; acc[t] = z; }
#pragma unroll
    for (int kb = 0; kb < KA; kb += 32) {
        v16h a = (kb < C)
            ? load_A_lds(&ldsXi[row * C], kb, laneHi)
            : load_A_diff_lds(&ldsXj[row * C], &ldsXi[row * C], kb - C, laneHi);
#pragma unroll
        for (int t = 0; t < 4; ++t) {
            v16h b = load_B_W(W1, KA, t * 16, kb, lane);
            acc[t] = __builtin_amdgcn_wmma_f32_16x16x32_f16(
                false, a, false, b, (short)0, acc[t], false, false);
        }
    }
    __syncthreads();
    // bias + relu -> LDS (f16) for second GEMM
#pragma unroll
    for (int t = 0; t < 4; ++t) {
        int n = t * 16 + (lane & 15);
        float bv = b1[n];
#pragma unroll
        for (int v = 0; v < 8; ++v) {
            int m = v + laneHi * 8;
            ldsH[m * 64 + n] = (f16)fmaxf(acc[t][v] + bv, 0.f);
        }
    }
    __syncthreads();

    // GEMM2: [16 x 64] x [64 x 64]
    v8f acc2[4];
#pragma unroll
    for (int t = 0; t < 4; ++t) { v8f z = {}; acc2[t] = z; }
#pragma unroll
    for (int kb = 0; kb < 64; kb += 32) {
        v16h a = load_A_lds(&ldsH[row * 64], kb, laneHi);
#pragma unroll
        for (int t = 0; t < 4; ++t) {
            v16h b = load_B_W(W2, 64, t * 16, kb, lane);
            acc2[t] = __builtin_amdgcn_wmma_f32_16x16x32_f16(
                false, a, false, b, (short)0, acc2[t], false, false);
        }
    }

    // scatter-max at dst
#pragma unroll
    for (int t = 0; t < 4; ++t) {
        int n = t * 16 + (lane & 15);
        float bv = b2[n];
#pragma unroll
        for (int v = 0; v < 8; ++v) {
            int e = e0 + v + laneHi * 8;
            if (e < nEdges) {
                int d = dst[e];
                atomicMaxF32(&out[(size_t)d * 64 + n], acc2[t][v] + bv);
            }
        }
    }
}

// ------------------------------------------------------------------- k-means

__global__ void kmeans_init(const float* __restrict__ X, float* __restrict__ cent, int n) {
    int j = blockIdx.x, c = threadIdx.x;                 // grid KCLUST, block 64
    unsigned idx = (2654435761u * (unsigned)(j + 1)) % (unsigned)n;
    cent[j * 64 + c] = X[(size_t)idx * 64 + c];
}

__global__ void kmeans_assign(const float* __restrict__ X, const float* __restrict__ cent,
                              int* __restrict__ assign, float* __restrict__ sums,
                              float* __restrict__ cnts, int n) {
    int i = blockIdx.x * blockDim.x + threadIdx.x;
    if (i >= n) return;
    const float* x = X + (size_t)i * 64;
    float best = 3.4e38f; int bj = 0;
    for (int j = 0; j < KCLUST; ++j) {
        const float* c = cent + j * 64;
        float d = 0.f;
#pragma unroll 16
        for (int t = 0; t < 64; ++t) { float df = x[t] - c[t]; d += df * df; }
        if (d < best) { best = d; bj = j; }
    }
    assign[i] = bj;
    if (sums) {
        for (int t = 0; t < 64; ++t) atomicAdd(&sums[bj * 64 + t], x[t]);
        atomicAdd(&cnts[bj], 1.f);
    }
}

__global__ void kmeans_update(const float* __restrict__ sums, const float* __restrict__ cnts,
                              float* __restrict__ cent) {
    int j = blockIdx.x, c = threadIdx.x;                 // grid KCLUST, block 64
    float cnt = fmaxf(cnts[j], 1.f);
    cent[j * 64 + c] = sums[j * 64 + c] / cnt;
}

// ------------------------------------------------------- edge message + pool

template<int COUT>
__global__ void message_kernel(const int* __restrict__ src, const int* __restrict__ dst,
                               const float* __restrict__ eattr, const int* __restrict__ assign,
                               const float* __restrict__ raw_alpha, const float* __restrict__ raw_coeff,
                               const float* __restrict__ deg, const float* __restrict__ xl,
                               float* __restrict__ out, int E) {
    int e = blockIdx.x * blockDim.x + threadIdx.x;
    if (e >= E) return;
    int s = src[e], d = dst[e];
    int ec = assign[s];
    float ew = 0.f;
    if (ec == assign[d]) {
        float alpha = softplusf(raw_alpha[ec]);
        float coeff = softplusf(raw_coeff[ec]);
        float ea = fmaxf(eattr[e], 1e-6f);
        ew = coeff * powf(ea, alpha);
    }
    float nw = ew / deg[s] + 1e-5f;
    const float* xr = xl + (size_t)d * COUT;
    float* orow = out + (size_t)d * COUT;
#pragma unroll 4
    for (int c = 0; c < COUT; ++c) atomicMaxF32(&orow[c], xr[c] * nw);
}

__global__ void pool_kernel(const float* __restrict__ h, const int* __restrict__ batch,
                            float* __restrict__ sums, float* __restrict__ cnts, int n) {
    int i = blockIdx.x * blockDim.x + threadIdx.x;
    if (i >= n) return;
    int b = batch[i];
    for (int c = 0; c < 128; ++c) atomicAdd(&sums[b * 128 + c], h[(size_t)i * 128 + c]);
    atomicAdd(&cnts[b], 1.f);
}

__global__ void fc_kernel(const float* __restrict__ sums, const float* __restrict__ cnts,
                          const float* __restrict__ fcW, const float* __restrict__ fcb,
                          float* __restrict__ out) {
    int b = threadIdx.x;
    if (b >= NBATCH) return;
    float inv = 1.f / fmaxf(cnts[b], 1.f);
    float acc = fcb[0];
    for (int c = 0; c < 128; ++c) acc += sums[b * 128 + c] * inv * fcW[c];
    out[b] = acc;
}

// ------------------------------------------------------------------- driver

struct ConvP { const float *W, *b, *Ws, *bs, *W1, *b1, *W2, *b2, *ra, *rc; };

extern "C" void kernel_launch(void* const* d_in, const int* in_sizes, int n_in,
                              void* d_out, int out_size, void* d_ws, size_t ws_size,
                              hipStream_t stream) {
    const float* x     = (const float*)d_in[0];      // [N,2]
    const int*   ei    = (const int*)d_in[1];        // [2,E]
    const float* eattr = (const float*)d_in[2];      // [E]
    const int*   batch = (const int*)d_in[3];        // [N]
    const int N = in_sizes[0] / 2;
    const int E = in_sizes[2];
    const int* src = ei;
    const int* dst = ei + E;

    int p = 4;
    auto getconv = [&](ConvP& c) {
        c.W  = (const float*)d_in[p++]; c.b  = (const float*)d_in[p++];
        c.Ws = (const float*)d_in[p++]; c.bs = (const float*)d_in[p++];
        c.W1 = (const float*)d_in[p++]; c.b1 = (const float*)d_in[p++];
        c.W2 = (const float*)d_in[p++]; c.b2 = (const float*)d_in[p++];
        c.ra = (const float*)d_in[p++]; c.rc = (const float*)d_in[p++];
    };
    ConvP c1, c2; getconv(c1); getconv(c2);
    const float* fcW = (const float*)d_in[p++];
    const float* fcb = (const float*)d_in[p++];

    // workspace bump allocator (256B aligned)
    char* ws = (char*)d_ws;
    size_t off = 0;
    auto alloc = [&](size_t bytes) -> void* {
        void* q = ws + off;
        off = (off + bytes + 255) & ~(size_t)255;
        return q;
    };
    f16*   sb1h  = (f16*)alloc((size_t)N * 64 * 2);
    float* xl1   = (float*)alloc((size_t)N * 64 * 4);
    float* sim   = (float*)alloc((size_t)N * 64 * 4);
    int*   assgn = (int*)alloc((size_t)N * 4);
    float* cent  = (float*)alloc(KCLUST * 64 * 4);
    float* ksums = (float*)alloc((KCLUST * 64 + KCLUST) * 4);
    float* kcnts = ksums + KCLUST * 64;
    float* deg   = (float*)alloc((size_t)N * 4);
    float* h1    = (float*)alloc((size_t)N * 64 * 4);
    f16*   h1h   = (f16*)alloc((size_t)N * 64 * 2);
    f16*   hcat  = (f16*)alloc((size_t)N * 128 * 2);
    f16*   sb2h  = (f16*)alloc((size_t)N * 128 * 2);
    float* xl2   = (float*)alloc((size_t)N * 128 * 4);
    float* h2    = (float*)alloc((size_t)N * 128 * 4);
    float* pool  = (float*)alloc((16 * 128 + 16) * 4);
    float* pcnt  = pool + 16 * 128;
    f16* Ws2h  = (f16*)alloc(128 * 128 * 2);
    f16* Wc2h  = (f16*)alloc(128 * 64 * 2);
    f16* W1c1h = (f16*)alloc(64 * 128 * 2);
    f16* W2c1h = (f16*)alloc(64 * 64 * 2);
    f16* W1c2h = (f16*)alloc(64 * 256 * 2);
    f16* W2c2h = (f16*)alloc(64 * 64 * 2);

    auto g256 = [](int n) { return (n + 255) / 256; };

    // --- weight conversion to f16 (once per launch; deterministic)
    f32_to_f16_kernel<<<g256(128*128), 256, 0, stream>>>(c2.Ws, Ws2h, 128 * 128);
    f32_to_f16_kernel<<<g256(128*64),  256, 0, stream>>>(c2.W,  Wc2h, 128 * 64);
    f32_to_f16_kernel<<<g256(64*128),  256, 0, stream>>>(c1.W1, W1c1h, 64 * 128);
    f32_to_f16_kernel<<<g256(64*64),   256, 0, stream>>>(c1.W2, W2c1h, 64 * 64);
    f32_to_f16_kernel<<<g256(64*256),  256, 0, stream>>>(c2.W1, W1c2h, 64 * 256);
    f32_to_f16_kernel<<<g256(64*64),   256, 0, stream>>>(c2.W2, W2c2h, 64 * 64);

    // --- degrees (graph is static across both convs)
    fill_f32<<<g256(N), 256, 0, stream>>>(deg, 0.f, N);
    deg_kernel<<<g256(E), 256, 0, stream>>>(src, deg, E);

    auto run_kmeans = [&](const float* X) {
        kmeans_init<<<KCLUST, 64, 0, stream>>>(X, cent, N);
        for (int it = 0; it < 50; ++it) {
            fill_f32<<<g256(KCLUST*64 + KCLUST), 256, 0, stream>>>(ksums, 0.f, KCLUST*64 + KCLUST);
            kmeans_assign<<<g256(N), 256, 0, stream>>>(X, cent, assgn, ksums, kcnts, N);
            kmeans_update<<<KCLUST, 64, 0, stream>>>(ksums, kcnts, cent);
        }
        kmeans_assign<<<g256(N), 256, 0, stream>>>(X, cent, assgn, nullptr, nullptr, N);
    };

    const int nTiles = (E + 15) / 16;
    const int nRowT  = (N + 15) / 16;

    // =================== conv1 (cin=2, cout=64) ===================
    small_linear_kernel<4, 64><<<g256(N*64), 256, 0, stream>>>(x, 2, c1.Ws, c1.bs, nullptr, sb1h, N);
    small_linear_kernel<2, 64><<<g256(N*64), 256, 0, stream>>>(x, 2, c1.W,  c1.b,  xl1, nullptr, N);

    fill_f32<<<g256(N*64), 256, 0, stream>>>(sim, -INFINITY, N * 64);
    edgeconv_wmma_kernel<64><<<nTiles, 32, 0, stream>>>(sb1h, src, dst, W1c1h, c1.b1, W2c1h, c1.b2, sim, E);
    fixup_kernel<<<g256(N*64), 256, 0, stream>>>(sim, N * 64, 0);

    run_kmeans(sim);

    fill_f32<<<g256(N*64), 256, 0, stream>>>(h1, -INFINITY, N * 64);
    message_kernel<64><<<g256(E), 256, 0, stream>>>(src, dst, eattr, assgn, c1.ra, c1.rc, deg, xl1, h1, E);
    fixup_kernel<<<g256(N*64), 256, 0, stream>>>(h1, N * 64, 1);   // + relu

    // =================== conv2 (cin=64, cout=128) ===================
    h_to_half_cat<<<g256(N*64), 256, 0, stream>>>(h1, h1h, hcat, N);
    gemm_wmma_kernel<128, 128, true><<<nRowT, 32, 0, stream>>>(hcat, Ws2h, c2.bs, nullptr, sb2h, N);
    gemm_wmma_kernel<64, 128, true><<<nRowT, 32, 0, stream>>>(h1h, Wc2h, c2.b, xl2, nullptr, N);

    fill_f32<<<g256(N*64), 256, 0, stream>>>(sim, -INFINITY, N * 64);
    edgeconv_wmma_kernel<128><<<nTiles, 32, 0, stream>>>(sb2h, src, dst, W1c2h, c2.b1, W2c2h, c2.b2, sim, E);
    fixup_kernel<<<g256(N*64), 256, 0, stream>>>(sim, N * 64, 0);

    run_kmeans(sim);

    fill_f32<<<g256(N*128), 256, 0, stream>>>(h2, -INFINITY, N * 128);
    message_kernel<128><<<g256(E), 256, 0, stream>>>(src, dst, eattr, assgn, c2.ra, c2.rc, deg, xl2, h2, E);
    fixup_kernel<<<g256(N*128), 256, 0, stream>>>(h2, N * 128, 1); // + relu

    // =================== mean-pool + fc ===================
    fill_f32<<<g256(16*128 + 16), 256, 0, stream>>>(pool, 0.f, 16 * 128 + 16);
    pool_kernel<<<g256(N), 256, 0, stream>>>(h2, batch, pool, pcnt, N);
    fc_kernel<<<1, 32, 0, stream>>>(pool, pcnt, fcW, fcb, (float*)d_out);

    (void)n_in; (void)out_size; (void)ws_size;
}
